// Informer_76665166233835
// MI455X (gfx1250) — matrix operand order repaired and checked by reading, not compiled, not run on hardware
//
#include <hip/hip_runtime.h>
#include <cmath>

#define B_      8
#define L_ENC_  2048
#define ENC_IN_ 7
#define D_      512
#define H_      8
#define DK_     64
#define DFF_    2048
#define TGT_    1024
#define COUT_   7
#define MAXU_   64
#define FFN_CHUNK 2048

#define TM 128
#define TN 128
#define TK 32

typedef __attribute__((ext_vector_type(16))) _Float16 v16h;
typedef __attribute__((ext_vector_type(8)))  _Float16 v8h;
typedef __attribute__((ext_vector_type(4)))  _Float16 v4h;
typedef __attribute__((ext_vector_type(8)))  float    v8f;
typedef __attribute__((ext_vector_type(4)))  float    v4f;

// ---------------------------------------------------------------------------
// Tiled f32->f16 WMMA GEMM:  C[M,N] = A[M,K] @ B[K,N] (+ bias[N])
// Block tile 128x128, 8 waves in 4(M)x2(N) grid; each wave computes 32x64 via
// EIGHT v_wmma_f32_16x16x32_f16 per 32-wide K step (2 A frags x 4 B frags).
// LDS double-buffered; next tile's global loads issued before compute.
// All LDS staging stores are packed v4h (ds_store_b64).
// M % 128 == 0, N % 128 == 0, K % 32 == 0.
// ---------------------------------------------------------------------------
__global__ __launch_bounds__(256) void k_gemm_wmma(
    const float* __restrict__ A, const float* __restrict__ Bm,
    const float* __restrict__ bias, float* __restrict__ C,
    int M, int N, int K)
{
  __shared__ __align__(64) _Float16 sA[2][TM][TK];   // [m][k]  2 x 8 KB
  __shared__ __align__(64) _Float16 sB[2][TN][TK];   // [n][k]  2 x 8 KB (transposed)
  const int tid  = threadIdx.x;
  const int wave = tid >> 5;
  const int lane = tid & 31;
  const int m0 = blockIdx.y * TM;
  const int n0 = blockIdx.x * TN;
  const int wm = (wave & 3) * 32;     // 4 waves along M
  const int wn = (wave >> 2) * 64;    // 2 waves along N

  v4f rA[4], rB[4];
  const int kqB = (tid >> 5) << 2;    // B staging: k-quad 0..28
  const int c4B = (tid & 31) << 2;    // B staging: n-quad 0..124

  // Issue global loads for the tile starting at k0 (held in registers).
  auto loadT = [&](int k0) {
#pragma unroll
    for (int it = 0; it < 4; ++it) {               // A: 128x32 = 1024 float4
      int i = tid + it * 256;
      int r = i >> 3, c = (i & 7) << 2;
      rA[it] = *(const v4f*)(A + (size_t)(m0 + r) * K + k0 + c);
    }
#pragma unroll
    for (int j = 0; j < 4; ++j)                    // B: rows k..k+3, same n-quad
      rB[j] = *(const v4f*)(Bm + (size_t)(k0 + kqB + j) * N + n0 + c4B);
  };
  // Convert registers to f16 and commit into LDS buffer `buf` (b64 stores).
  auto storeT = [&](int buf) {
#pragma unroll
    for (int it = 0; it < 4; ++it) {
      int i = tid + it * 256;
      int r = i >> 3, c = (i & 7) << 2;
      v4h p;
      p[0] = (_Float16)rA[it][0]; p[1] = (_Float16)rA[it][1];
      p[2] = (_Float16)rA[it][2]; p[3] = (_Float16)rA[it][3];
      *(v4h*)&sA[buf][r][c] = p;
    }
#pragma unroll
    for (int j = 0; j < 4; ++j) {                  // transpose: k..k+3 contiguous
      v4h p;
      p[0] = (_Float16)rB[0][j]; p[1] = (_Float16)rB[1][j];
      p[2] = (_Float16)rB[2][j]; p[3] = (_Float16)rB[3][j];
      *(v4h*)&sB[buf][c4B + j][kqB] = p;
    }
  };

  v8f acc[2][4];
#pragma unroll
  for (int i = 0; i < 2; ++i)
#pragma unroll
    for (int j = 0; j < 4; ++j) acc[i][j] = (v8f){};

  loadT(0);
  storeT(0);
  __syncthreads();

  int buf = 0;
  for (int k0 = 0; k0 < K; k0 += TK) {
    const bool more = (k0 + TK) < K;
    if (more) loadT(k0 + TK);                      // hide global latency behind math

    // A fragments (ISA 16-bit A layout): lane half selects K-base 0/8 (+16).
    const int kbA = (lane < 16) ? 0 : 8;
    const int kbB = (lane < 16) ? 0 : 8;
    v16h a[2], bf[4];
#pragma unroll
    for (int t2 = 0; t2 < 2; ++t2) {
      int mr = wm + (lane & 15) + t2 * 16;
      v8h lo = *(const v8h*)&sA[buf][mr][kbA];
      v8h hi = *(const v8h*)&sA[buf][mr][kbA + 16];
#pragma unroll
      for (int i = 0; i < 8; ++i) { a[t2][i] = lo[i]; a[t2][i + 8] = hi[i]; }
    }
#pragma unroll
    for (int j = 0; j < 4; ++j) {                  // same K split as A: kbB, kbB+16
      v8h lo = *(const v8h*)&sB[buf][wn + j * 16 + (lane & 15)][kbB];
      v8h hi = *(const v8h*)&sB[buf][wn + j * 16 + (lane & 15)][kbB + 16];
      for (int i = 0; i < 8; ++i) { bf[j][i] = lo[i]; bf[j][i + 8] = hi[i]; }
    }

#pragma unroll
    for (int t2 = 0; t2 < 2; ++t2)
#pragma unroll
      for (int j = 0; j < 4; ++j)
        acc[t2][j] = __builtin_amdgcn_wmma_f32_16x16x32_f16(
            false, a[t2], false, bf[j], (short)0, acc[t2][j], false, false);

    if (more) storeT(buf ^ 1);                     // fill the other buffer
    __syncthreads();
    buf ^= 1;
  }

  // D layout: VGPR r -> M = r + 8*(lane>=16), N = lane%16.
  const int rb = m0 + wm + ((lane & 16) ? 8 : 0);
  const int cb = n0 + wn + (lane & 15);
#pragma unroll
  for (int t2 = 0; t2 < 2; ++t2) {
#pragma unroll
    for (int j = 0; j < 4; ++j) {
      const float bb = bias ? bias[cb + j * 16] : 0.f;
#pragma unroll
      for (int r = 0; r < 8; ++r)
        C[(size_t)(rb + t2 * 16 + r) * N + cb + j * 16] = acc[t2][j][r] + bb;
    }
  }
}

// ---------------------------------------------------------------------------
// LayerNorm over last dim (D_=512), optional residual: out = LN(x [+ res])*g+b
// One wave per row (wave32 shfl reduction).
// ---------------------------------------------------------------------------
__global__ __launch_bounds__(256) void k_layernorm(
    const float* __restrict__ x, const float* __restrict__ res,
    const float* __restrict__ g, const float* __restrict__ bta,
    float* __restrict__ o, int rows)
{
  int row = blockIdx.x * 8 + (threadIdx.x >> 5);
  if (row >= rows) return;
  int lane = threadIdx.x & 31;
  const float* xr = x + (size_t)row * D_;
  const float* rr = res ? res + (size_t)row * D_ : nullptr;
  float s = 0.f, s2 = 0.f;
  for (int i = lane; i < D_; i += 32) {
    float v = xr[i] + (rr ? rr[i] : 0.f);
    s += v; s2 += v * v;
  }
#pragma unroll
  for (int off = 16; off; off >>= 1) {
    s  += __shfl_xor(s,  off, 32);
    s2 += __shfl_xor(s2, off, 32);
  }
  float mean = s * (1.f / D_);
  float var  = s2 * (1.f / D_) - mean * mean;
  float inv  = rsqrtf(var + 1e-5f);
  float* orow = o + (size_t)row * D_;
  for (int i = lane; i < D_; i += 32) {
    float v = xr[i] + (rr ? rr[i] : 0.f);
    orow[i] = (v - mean) * inv * g[i] + bta[i];
  }
}

// Token embedding: width-3 conv over circularly padded sequence, 'WIO' weights
__global__ void k_conv_embed(const float* __restrict__ x, const float* __restrict__ w,
                             float* __restrict__ out, int Bn, int L, int Cin)
{
  size_t idx = (size_t)blockIdx.x * blockDim.x + threadIdx.x;
  size_t total = (size_t)Bn * L * D_;
  if (idx >= total) return;
  int dch = (int)(idx % D_);
  int t   = (int)((idx / D_) % L);
  int bb  = (int)(idx / ((size_t)D_ * L));
  float acc = 0.f;
#pragma unroll
  for (int kw = 0; kw < 3; ++kw) {
    int ti = t + kw - 1;
    ti = (ti < 0) ? (L - 1) : ((ti >= L) ? 0 : ti);
    const float* xp = x + ((size_t)bb * L + ti) * Cin;
    const float* wp = w + (size_t)kw * Cin * D_ + dch;
    for (int c = 0; c < Cin; ++c) acc += xp[c] * wp[(size_t)c * D_];
  }
  out[idx] = acc;
}

__global__ void k_gelu(float* __restrict__ x, size_t n)
{
  size_t i = (size_t)blockIdx.x * blockDim.x + threadIdx.x;
  if (i >= n) return;
  float v = x[i];
  x[i] = 0.5f * v * (1.f + erff(v * 0.70710678118654752f));
}

__global__ void k_fill(float* __restrict__ x, float v, size_t n)
{
  size_t i = (size_t)blockIdx.x * blockDim.x + threadIdx.x;
  if (i < n) x[i] = v;
}

// Sparsity measure M = max_u(q.k_u) - sum_u(q.k_u)/Lk over U sampled keys
__global__ void k_sample_m(const float* __restrict__ q, const float* __restrict__ k,
                           float* __restrict__ Mout, int Lq, int Lk, int U, unsigned seed)
{
  int idx = blockIdx.x * blockDim.x + threadIdx.x;       // over B*H*Lq
  if (idx >= B_ * H_ * Lq) return;
  int qi = idx % Lq;
  int h  = (idx / Lq) % H_;
  int bb = idx / (Lq * H_);
  const float* qp = q + ((size_t)bb * Lq + qi) * D_ + h * DK_;
  float mx = -INFINITY, sum = 0.f;
  for (int u = 0; u < U; ++u) {
    unsigned r = (unsigned)(qi * U + u) * 2654435761u + seed * 0x9E3779B9u;
    r ^= r >> 13; r *= 0x85EBCA6Bu; r ^= r >> 16;
    int kk = (int)(r % (unsigned)Lk);
    const float* kp = k + ((size_t)bb * Lk + kk) * D_ + h * DK_;
    float dot = 0.f;
    for (int e = 0; e < DK_; ++e) dot += qp[e] * kp[e];
    mx = fmaxf(mx, dot); sum += dot;
  }
  Mout[idx] = mx - sum / (float)Lk;
}

// Top-u indices of M per (b,h); destructive on Mrow (scratch).
__global__ __launch_bounds__(256) void k_topk(float* __restrict__ Mrow,
                                              int* __restrict__ top, int Lq, int u)
{
  int bh = blockIdx.x;
  float* m = Mrow + (size_t)bh * Lq;
  int* t = top + bh * MAXU_;
  __shared__ float sv[256];
  __shared__ int   si[256];
  for (int it = 0; it < u; ++it) {
    float best = -INFINITY; int bi = 0;
    for (int i = threadIdx.x; i < Lq; i += 256) {
      float v = m[i];
      if (v > best) { best = v; bi = i; }
    }
    sv[threadIdx.x] = best; si[threadIdx.x] = bi;
    __syncthreads();
    for (int s = 128; s; s >>= 1) {
      if (threadIdx.x < s && sv[threadIdx.x + s] > sv[threadIdx.x]) {
        sv[threadIdx.x] = sv[threadIdx.x + s];
        si[threadIdx.x] = si[threadIdx.x + s];
      }
      __syncthreads();
    }
    if (threadIdx.x == 0) { t[it] = si[0]; m[si[0]] = -INFINITY; }
    __syncthreads();
  }
}

// Masked path: ctx = cumsum(V) along sequence (Lk == Lq).
__global__ void k_ctx_cumsum(const float* __restrict__ v, float* __restrict__ ctx, int Lq)
{
  int idx = blockIdx.x * blockDim.x + threadIdx.x;       // B*H*DK
  if (idx >= B_ * H_ * DK_) return;
  int e = idx % DK_; int h = (idx / DK_) % H_; int bb = idx / (DK_ * H_);
  float acc = 0.f;
  for (int l = 0; l < Lq; ++l) {
    size_t p = ((size_t)bb * Lq + l) * D_ + h * DK_ + e;
    acc += v[p];
    ctx[p] = acc;
  }
}

__global__ void k_vmean(const float* __restrict__ v, float* __restrict__ vm, int Lk)
{
  int idx = blockIdx.x * blockDim.x + threadIdx.x;       // B*H*DK
  if (idx >= B_ * H_ * DK_) return;
  int e = idx % DK_; int h = (idx / DK_) % H_; int bb = idx / (DK_ * H_);
  float acc = 0.f;
  for (int l = 0; l < Lk; ++l)
    acc += v[((size_t)bb * Lk + l) * D_ + h * DK_ + e];
  vm[(bb * H_ + h) * DK_ + e] = acc / (float)Lk;
}

__global__ void k_ctx_bcast(const float* __restrict__ vm, float* __restrict__ ctx, int Lq)
{
  size_t idx = (size_t)blockIdx.x * blockDim.x + threadIdx.x;  // B*Lq*D
  if (idx >= (size_t)B_ * Lq * D_) return;
  int col = (int)(idx % D_);
  int tok = (int)(idx / D_);
  int bb  = tok / Lq;
  ctx[idx] = vm[(bb * H_ + col / DK_) * DK_ + (col % DK_)];
}

// Softmax attention update for the u selected queries; one block per (b,h,ui)
__global__ __launch_bounds__(256) void k_attn_update(
    const float* __restrict__ q, const float* __restrict__ k,
    const float* __restrict__ v, const int* __restrict__ top,
    float* __restrict__ ctx, int Lq, int Lk, int u, int maskf)
{
  __shared__ float sc[2048];
  __shared__ float red[256];
  __shared__ float part[4][64];
  __shared__ float qs[64];
  int ui = blockIdx.x % u;
  int h  = (blockIdx.x / u) % H_;
  int bb = blockIdx.x / (u * H_);
  int qi = top[(bb * H_ + h) * MAXU_ + ui];
  const float* qp = q + ((size_t)bb * Lq + qi) * D_ + h * DK_;
  if (threadIdx.x < 64) qs[threadIdx.x] = qp[threadIdx.x];
  __syncthreads();

  float lmax = -INFINITY;
  for (int kk = threadIdx.x; kk < Lk; kk += 256) {
    const float* kp = k + ((size_t)bb * Lk + kk) * D_ + h * DK_;
    float dot = 0.f;
    for (int e = 0; e < DK_; ++e) dot += qs[e] * kp[e];
    dot *= 0.125f;                              // 1/sqrt(64)
    if (maskf && kk > qi) dot = -INFINITY;
    sc[kk] = dot;
    lmax = fmaxf(lmax, dot);
  }
  red[threadIdx.x] = lmax; __syncthreads();
  for (int s = 128; s; s >>= 1) {
    if (threadIdx.x < s) red[threadIdx.x] = fmaxf(red[threadIdx.x], red[threadIdx.x + s]);
    __syncthreads();
  }
  float gmax = red[0]; __syncthreads();

  float lsum = 0.f;
  for (int kk = threadIdx.x; kk < Lk; kk += 256) {
    float e = expf(sc[kk] - gmax);
    sc[kk] = e; lsum += e;
  }
  red[threadIdx.x] = lsum; __syncthreads();
  for (int s = 128; s; s >>= 1) {
    if (threadIdx.x < s) red[threadIdx.x] += red[threadIdx.x + s];
    __syncthreads();
  }
  float inv = 1.f / red[0];
  __syncthreads();

  int e = threadIdx.x & 63, chunk = threadIdx.x >> 6;
  int kq = Lk >> 2;
  float acc = 0.f;
  for (int kk = chunk * kq; kk < (chunk + 1) * kq; ++kk)
    acc += sc[kk] * v[((size_t)bb * Lk + kk) * D_ + h * DK_ + e];
  part[chunk][e] = acc; __syncthreads();
  if (threadIdx.x < 64) {
    float o = (part[0][threadIdx.x] + part[1][threadIdx.x] +
               part[2][threadIdx.x] + part[3][threadIdx.x]) * inv;
    ctx[((size_t)bb * Lq + qi) * D_ + h * DK_ + threadIdx.x] = o;
  }
}

// Final projection to COUT_=7 (too narrow for WMMA tiles; tiny anyway)
__global__ void k_proj(const float* __restrict__ d, const float* __restrict__ w,
                       const float* __restrict__ b, float* __restrict__ out, int rows)
{
  int idx = blockIdx.x * blockDim.x + threadIdx.x;
  if (idx >= rows * COUT_) return;
  int c = idx % COUT_, r = idx / COUT_;
  const float* dr = d + (size_t)r * D_;
  float acc = b[c];
  for (int i = 0; i < D_; ++i) acc += dr[i] * w[(size_t)i * COUT_ + c];
  out[idx] = acc;
}

// ---------------------------------------------------------------------------
extern "C" void kernel_launch(void* const* d_in, const int* in_sizes, int n_in,
                              void* d_out, int out_size, void* d_ws, size_t ws_size,
                              hipStream_t stream)
{
  (void)in_sizes; (void)n_in; (void)out_size; (void)ws_size;
  const int NE = B_ * L_ENC_;   // 16384 encoder tokens
  const int ND = B_ * TGT_;     // 8192 decoder tokens

  // --- input pointer map (setup_inputs insertion-order flatten) ---
  const float* x_enc       = (const float*)d_in[0];
  const float* embed_enc_w = (const float*)d_in[1];
  // d_in[2] = embed_dec_w (conv of zeros -> unused)
  const int ENC_BASE = 3;       // + 16*layer : wq bq wk bk wv bv wo bo w1 b1 w2 b2 ln1g ln1b ln2g ln2b
  const int DEC_BASE = 3 + 16 * 2; // s_(8) c_(8) w1 b1 w2 b2 ln1g..ln3b
  const float* enc_ng = (const float*)d_in[61];
  const float* enc_nb = (const float*)d_in[62];
  const float* dec_ng = (const float*)d_in[63];
  const float* dec_nb = (const float*)d_in[64];
  const float* proj_w = (const float*)d_in[65];
  const float* proj_b = (const float*)d_in[66];

  // --- workspace carve (256B aligned) ---
  char* wsp = (char*)d_ws;
  auto alloc = [&](size_t nfloats) -> float* {
    float* p = (float*)wsp;
    wsp += ((nfloats * sizeof(float) + 255) / 256) * 256;
    return p;
  };
  float* X   = alloc((size_t)NE * D_);
  float* Q   = alloc((size_t)NE * D_);
  float* Kb  = alloc((size_t)NE * D_);
  float* Vb  = alloc((size_t)NE * D_);
  float* CTX = alloc((size_t)NE * D_);
  float* Y   = alloc((size_t)NE * D_);
  float* ENC = alloc((size_t)NE * D_);
  float* Hb  = alloc((size_t)FFN_CHUNK * DFF_);
  float* Mb  = alloc((size_t)B_ * H_ * L_ENC_);
  int*   TOP = (int*)alloc((size_t)B_ * H_ * MAXU_);
  float* VM  = alloc((size_t)B_ * H_ * DK_);

  auto gemm = [&](const float* A, const float* Bm, const float* bias, float* C,
                  int M, int N, int K) {
    dim3 g((unsigned)(N / TN), (unsigned)(M / TM));
    k_gemm_wmma<<<g, 256, 0, stream>>>(A, Bm, bias, C, M, N, K);
  };
  auto lnorm = [&](const float* x, const float* res, const float* g,
                   const float* b, float* o, int rows) {
    k_layernorm<<<rows / 8, 256, 0, stream>>>(x, res, g, b, o, rows);
  };
  auto ufac = [](int L) {
    int v = 5 * (int)std::ceil(std::log((double)L));
    return v < L ? v : L;
  };
  auto ffn = [&](float* xin, const float* w1, const float* b1,
                 const float* w2, const float* b2, float* yout, int rows) {
    for (int m0 = 0; m0 < rows; m0 += FFN_CHUNK) {
      gemm(xin + (size_t)m0 * D_, w1, b1, Hb, FFN_CHUNK, DFF_, D_);
      size_t hn = (size_t)FFN_CHUNK * DFF_;
      k_gelu<<<(unsigned)((hn + 255) / 256), 256, 0, stream>>>(Hb, hn);
      gemm(Hb, w2, b2, yout + (size_t)m0 * D_, FFN_CHUNK, D_, DFF_);
    }
  };
  auto attn = [&](const float* xq, const float* xkv, int base, int Lq, int Lk,
                  int maskf, unsigned seed, float* outb) {
    const float* wq = (const float*)d_in[base + 0];
    const float* bq = (const float*)d_in[base + 1];
    const float* wk = (const float*)d_in[base + 2];
    const float* bk = (const float*)d_in[base + 3];
    const float* wv = (const float*)d_in[base + 4];
    const float* bv = (const float*)d_in[base + 5];
    const float* wo = (const float*)d_in[base + 6];
    const float* bo = (const float*)d_in[base + 7];
    gemm(xq,  wq, bq, Q,  B_ * Lq, D_, D_);
    gemm(xkv, wk, bk, Kb, B_ * Lk, D_, D_);
    gemm(xkv, wv, bv, Vb, B_ * Lk, D_, D_);
    int U = ufac(Lk), u = ufac(Lq);
    int tq = B_ * H_ * Lq;
    k_sample_m<<<(tq + 255) / 256, 256, 0, stream>>>(Q, Kb, Mb, Lq, Lk, U, seed);
    k_topk<<<B_ * H_, 256, 0, stream>>>(Mb, TOP, Lq, u);
    int tc = B_ * H_ * DK_;
    if (maskf) {
      k_ctx_cumsum<<<(tc + 255) / 256, 256, 0, stream>>>(Vb, CTX, Lq);
    } else {
      k_vmean<<<(tc + 255) / 256, 256, 0, stream>>>(Vb, VM, Lk);
      size_t tn = (size_t)B_ * Lq * D_;
      k_ctx_bcast<<<(unsigned)((tn + 255) / 256), 256, 0, stream>>>(VM, CTX, Lq);
    }
    k_attn_update<<<B_ * H_ * u, 256, 0, stream>>>(Q, Kb, Vb, TOP, CTX, Lq, Lk, u, maskf);
    gemm(CTX, wo, bo, outb, B_ * Lq, D_, D_);
  };

  unsigned cnt = 0;

  // ---------------- Encoder ----------------
  {
    size_t tn = (size_t)NE * D_;
    k_conv_embed<<<(unsigned)((tn + 255) / 256), 256, 0, stream>>>(
        x_enc, embed_enc_w, X, B_, L_ENC_, ENC_IN_);
  }
  for (int l = 0; l < 2; ++l) {
    int base = ENC_BASE + 16 * l;
    attn(X, X, base, L_ENC_, L_ENC_, 0, cnt++, Y);
    lnorm(X, Y, (const float*)d_in[base + 12], (const float*)d_in[base + 13], X, NE);
    ffn(X, (const float*)d_in[base + 8], (const float*)d_in[base + 9],
           (const float*)d_in[base + 10], (const float*)d_in[base + 11], Y, NE);
    lnorm(X, Y, (const float*)d_in[base + 14], (const float*)d_in[base + 15], X, NE);
  }
  lnorm(X, nullptr, enc_ng, enc_nb, ENC, NE);

  // ---------------- Decoder ----------------
  {
    size_t tn = (size_t)ND * D_;
    k_fill<<<(unsigned)((tn + 255) / 256), 256, 0, stream>>>(X, 0.f, tn); // embed of zeros
  }
  {
    int base = DEC_BASE;
    attn(X, X, base + 0, TGT_, TGT_, 1, cnt++, Y);           // masked self-attn
    lnorm(X, Y, (const float*)d_in[base + 20], (const float*)d_in[base + 21], X, ND);
    attn(X, ENC, base + 8, TGT_, L_ENC_, 0, cnt++, Y);       // cross-attn
    lnorm(X, Y, (const float*)d_in[base + 22], (const float*)d_in[base + 23], X, ND);
    ffn(X, (const float*)d_in[base + 16], (const float*)d_in[base + 17],
           (const float*)d_in[base + 18], (const float*)d_in[base + 19], Y, ND);
    lnorm(X, Y, (const float*)d_in[base + 24], (const float*)d_in[base + 25], X, ND);
  }
  lnorm(X, nullptr, dec_ng, dec_nb, X, ND);

  k_proj<<<(ND * COUT_ + 255) / 256, 256, 0, stream>>>(X, proj_w, proj_b,
                                                       (float*)d_out, ND);
}
